// GraphNativeEncoder_695784702510
// MI455X (gfx1250) — compile-verified, matrix operand order checked by reading, not run on hardware
//
#include <hip/hip_runtime.h>
#include <cstdint>

// Problem constants (from reference): N=4096, T=32, C=H=64, K=10
#define TT 32
#define CC 64
#define HH 64
#define KTOP 10
#define SLOPE 0.2f
#define CHUNK 64
#define PADR 38   // 3 halo + 32 + 3 halo rows per node (covers dilation-3 conv)
#define PADO 3

typedef __bf16 v16bf __attribute__((ext_vector_type(16)));
typedef float  v8f   __attribute__((ext_vector_type(8)));

__device__ __forceinline__ float sigm(float v){ return 1.0f/(1.0f + __expf(-v)); }

// float -> bf16 bits, round-to-nearest-even
__device__ __forceinline__ unsigned short f2bf(float f){
  unsigned u = __float_as_uint(f);
  u += 0x7fffu + ((u >> 16) & 1u);
  return (unsigned short)(u >> 16);
}

// Load one 16x32 bf16 WMMA fragment (A layout; B uses the same (row,k) indexing
// when weights are pre-transposed). rowptr points at (this lane's row, k=0).
// ISA 7.12.2: lanes 0-15 hold K={0,2,4,6,16,18,20,22}(+1); lanes 16-31 K +8.
// Unconditional: each half is 16B contiguous -> compiler emits 2x b128.
__device__ __forceinline__ v16bf frag_load(const unsigned short* rowptr, int lane){
  union { v16bf v; unsigned u[8]; } f;
  int half = lane >> 4;
#pragma unroll
  for (int q = 0; q < 8; ++q){
    int kb = (q < 4) ? (half*8 + q*2) : (16 + half*8 + (q-4)*2);
    f.u[q] = *(const unsigned*)(rowptr + kb);
  }
  return f.v;
}

// ---------------------------------------------------------------- weight prep
// Transpose + pre-scale weights into bf16, WMMA-B friendly (n-major rows of K).
__global__ void k_prep(const float* cfw, const float* cfb, const float* csw, const float* csb,
                       const float* sgate, const float* Wmsg, const float* Wself,
                       unsigned short* wconv, unsigned short* wmsgT, unsigned short* wselfT,
                       float* bcomb){
  float g = sigm(sgate[0]);
  int tid = threadIdx.x;
  // wconv[((cv*3+k)*64 + h)*64 + c] = scale * w[h,c,k]   (scale = g or 1-g)
  for (int idx = tid; idx < 2*3*64*64; idx += 256){
    int cv = idx / 12288, rem = idx % 12288;
    int k  = rem / 4096,  r2  = rem % 4096;
    int h  = r2 >> 6,     c   = r2 & 63;
    const float* w = (cv == 0) ? cfw : csw;
    float sc = (cv == 0) ? g : (1.0f - g);
    wconv[idx] = f2bf(sc * w[(h*64 + c)*3 + k]);
  }
  for (int idx = tid; idx < 4096; idx += 256){
    int n = idx >> 6, k = idx & 63;
    wmsgT[idx]  = f2bf(Wmsg [k*64 + n]);
    wselfT[idx] = f2bf(Wself[k*64 + n]);
  }
  if (tid < 64) bcomb[tid] = g*cfb[tid] + (1.0f - g)*csb[tid];
}

// x (N,32,64) f32 -> time-padded bf16 (N,38,64), halo rows zeroed.
__global__ void k_castpad(const float* x, unsigned short* xbp, int total){
  int i = blockIdx.x*blockDim.x + threadIdx.x;
  if (i >= total) return;
  int c = i & 63;
  int r = (i >> 6) % PADR;
  int node = i / (PADR*CC);
  int t = r - PADO;
  float v = ((unsigned)t < (unsigned)TT) ? x[((size_t)node*TT + t)*CC + c] : 0.f;
  xbp[i] = f2bf(v);
}

// ------------------------------------------------- x_agg -> projected, normed e
__global__ void k_meanproj(const float* x, const float* Wproj, float* e, int N){
  __shared__ float xagg[64];
  int n = blockIdx.x, tid = threadIdx.x;  // blockDim = 64
  const float* xr = x + (size_t)n*TT*CC;
  float s = 0.f;
  for (int t = 0; t < TT; ++t) s += xr[t*CC + tid];
  xagg[tid] = s * (1.0f/TT);
  __syncthreads();
  if (tid < 32){
    float v = 0.f;
    for (int c = 0; c < 64; ++c) v += xagg[c]*Wproj[c*32 + tid];
    float s2 = v*v;
    for (int off = 16; off >= 1; off >>= 1) s2 += __shfl_xor(s2, off);
    e[n*32 + tid] = v * (1.0f/(sqrtf(s2) + 1e-12f));
  }
}

// --------------------------------------------------------- per-row sim + top-K
__global__ void k_topk(const float* e, const float* mix_logit,
                       int* esrc, int* edst, float* eattr, int N, int E){
  __shared__ float srow[4096];
  __shared__ float ei[32];
  __shared__ float rv[256];
  __shared__ int   ri[256];
  int i = blockIdx.x, tid = threadIdx.x;
  if (tid < 32) ei[tid] = e[i*32 + tid];
  __syncthreads();
  for (int j = tid; j < N; j += 256){
    const float* ej = e + j*32;
    float s = 0.f;
#pragma unroll
    for (int p = 0; p < 32; ++p) s += ei[p]*ej[p];
    srow[j] = s;
  }
  __syncthreads();
  if (tid == 0) srow[i] = 3.0e38f;   // force self to be the dropped max
  __syncthreads();
  float alpha = sigm(mix_logit[0]);
  for (int it = 0; it <= KTOP; ++it){
    float bv = -3.4e38f; int bi = 0;
    for (int j = tid; j < N; j += 256) if (srow[j] > bv){ bv = srow[j]; bi = j; }
    rv[tid] = bv; ri[tid] = bi;
    __syncthreads();
    for (int off = 128; off >= 1; off >>= 1){
      if (tid < off && rv[tid+off] > rv[tid]){ rv[tid] = rv[tid+off]; ri[tid] = ri[tid+off]; }
      __syncthreads();
    }
    if (tid == 0){
      int b = ri[0];
      if (it > 0){
        int o = E + i*KTOP + (it - 1);
        esrc[o] = i; edst[o] = b; eattr[o] = rv[0]*alpha;
      }
      srow[b] = -3.4e38f;
    }
    __syncthreads();
  }
}

__global__ void k_fixed(const int* fei, const float* fea, const float* mix_logit,
                        int* esrc, int* edst, float* eattr, int E){
  int e = blockIdx.x*blockDim.x + threadIdx.x;
  if (e >= E) return;
  float alpha = sigm(mix_logit[0]);
  esrc[e] = fei[e];
  edst[e] = fei[E + e];
  eattr[e] = fea[e]*(1.0f - alpha);
}

// -------------------------------------- dual dilated conv as shifted WMMA GEMMs
// Input is the zero-padded bf16 buffer -> all fragment loads branch-free.
__global__ void k_conv(const unsigned short* xbp, const unsigned short* wconv, const float* bcomb,
                       float* xtf, unsigned short* xth, int N){
  int gtid = blockIdx.x*blockDim.x + threadIdx.x;
  int wid  = gtid >> 5, lane = threadIdx.x & 31;
  int node = wid >> 1, mt = wid & 1;
  if (node >= N) return;                    // uniform per wave (grid sized exactly)
  int t0 = mt*16;
  int m = lane & 15, half = lane >> 4, hn = lane & 15;
  const unsigned short* nodeBase = xbp + (size_t)node*PADR*CC + (PADO + t0 + m)*CC;
  v8f acc[4];
#pragma unroll
  for (int nt = 0; nt < 4; ++nt){
    float b = bcomb[nt*16 + hn];
#pragma unroll
    for (int r = 0; r < 8; ++r) acc[nt][r] = b;
  }
  const int dil[2] = {1, 3};
#pragma unroll
  for (int cv = 0; cv < 2; ++cv){
#pragma unroll
    for (int s = 0; s < 3; ++s){
      int shift = (s - 1)*dil[cv];
      const unsigned short* arow  = nodeBase + shift*CC;
      const unsigned short* wbase = wconv + (size_t)(cv*3 + s)*64*64;
#pragma unroll
      for (int ch = 0; ch < 2; ++ch){
        v16bf a = frag_load(arow + ch*32, lane);
#pragma unroll
        for (int nt = 0; nt < 4; ++nt){
          v16bf b = frag_load(wbase + (nt*16 + hn)*64 + ch*32, lane);
          acc[nt] = __builtin_amdgcn_wmma_f32_16x16x32_bf16(
              false, a, false, b, (short)0, acc[nt], false, false);
        }
      }
    }
  }
#pragma unroll
  for (int nt = 0; nt < 4; ++nt){
    int h = nt*16 + hn;
#pragma unroll
    for (int r = 0; r < 8; ++r){
      int t = t0 + r + 8*half;               // C layout: M = r + 8*(lane>=16)
      size_t off = ((size_t)node*TT + t)*HH + h;
      float v = acc[nt][r];
      xtf[off] = v;
      xth[off] = f2bf(v);
    }
  }
}

// --------------- generic per-row (rows x 64) @ (64 x 64) + bias, WMMA
// A addressing: node = row/32, physical row = node*aNodeStride + aRowOff + row%32.
__global__ void k_gemm(const unsigned short* A, int aNodeStride, int aRowOff,
                       const unsigned short* BT, const float* bias,
                       float* outF, int rows){
  int gtid = blockIdx.x*blockDim.x + threadIdx.x;
  int wid  = gtid >> 5, lane = threadIdx.x & 31;
  int row0 = wid*16;
  if (row0 >= rows) return;
  int node = row0 >> 5, tl = row0 & 31;
  int m = lane & 15, half = lane >> 4, hn = lane & 15;
  const unsigned short* abase = A + ((size_t)node*aNodeStride + aRowOff + tl + m)*64;
  v8f acc[4];
#pragma unroll
  for (int nt = 0; nt < 4; ++nt){
    float b = bias[nt*16 + hn];
#pragma unroll
    for (int r = 0; r < 8; ++r) acc[nt][r] = b;
  }
#pragma unroll
  for (int ch = 0; ch < 2; ++ch){
    v16bf a = frag_load(abase + ch*32, lane);
#pragma unroll
    for (int nt = 0; nt < 4; ++nt){
      v16bf b = frag_load(BT + (nt*16 + hn)*64 + ch*32, lane);
      acc[nt] = __builtin_amdgcn_wmma_f32_16x16x32_bf16(
          false, a, false, b, (short)0, acc[nt], false, false);
    }
  }
#pragma unroll
  for (int nt = 0; nt < 4; ++nt){
    int h = nt*16 + hn;
#pragma unroll
    for (int r = 0; r < 8; ++r){
      size_t off = (size_t)(row0 + r + 8*half)*64 + h;
      outF[off] = acc[nt][r];
    }
  }
}

// z bf16 copy for downstream reuse is not needed; but z scores are:
// ssrc = z . w_att_src (no bias), sdst = x_t . w_att_dst (no bias)
__global__ void k_scores(const float* z, const float* xt, const float* was, const float* wad,
                         float* ssrc, float* sdst, int rows){
  int i = blockIdx.x*blockDim.x + threadIdx.x;
  if (i >= rows) return;
  const float4* zr = (const float4*)(z  + (size_t)i*64);
  const float4* xr = (const float4*)(xt + (size_t)i*64);
  const float4* w1 = (const float4*)was;
  const float4* w2 = (const float4*)wad;
  float a = 0.f, b = 0.f;
#pragma unroll
  for (int q = 0; q < 16; ++q){
    float4 z4 = zr[q], x4 = xr[q], a4 = w1[q], b4 = w2[q];
    a += z4.x*a4.x + z4.y*a4.y + z4.z*a4.z + z4.w*a4.w;
    b += x4.x*b4.x + x4.y*b4.y + x4.z*b4.z + x4.w*b4.w;
  }
  ssrc[i] = a; sdst[i] = b;
}

// --------------------------------------------------------- CSR build (by dst)
__global__ void k_zero(int* a, int n){
  int i = blockIdx.x*blockDim.x + threadIdx.x;
  if (i < n) a[i] = 0;
}
__global__ void k_count(const int* edst, int* counts, int Etot){
  int e = blockIdx.x*blockDim.x + threadIdx.x;
  if (e < Etot) atomicAdd(&counts[edst[e]], 1);
}
__global__ void k_scan(const int* counts, int* offsets, int N){
  if (threadIdx.x == 0 && blockIdx.x == 0){
    int acc = 0; offsets[0] = 0;
    for (int i = 0; i < N; ++i){ acc += counts[i]; offsets[i+1] = acc; }
  }
}
__global__ void k_fill(const int* edst, const int* offsets, int* cursor, int* elist, int Etot){
  int e = blockIdx.x*blockDim.x + threadIdx.x;
  if (e >= Etot) return;
  int d = edst[e];
  int p = atomicAdd(&cursor[d], 1);
  elist[offsets[d] + p] = e;
}

// --------------- per-dst segment softmax + weighted aggregation (gather style)
__global__ void k_agg(const int* esrc, const float* eattr, const int* offsets, const int* elist,
                      const float* ssrc, const float* sdstv, const float* z,
                      const float* bas, const float* bad, float* out){
  __shared__ float sdl[32], smax[32], sden[32];
  __shared__ float accs[32*64];
  __shared__ float cb[CHUNK*32];
  __shared__ int   sb[CHUNK];
  __shared__ float ab[CHUNK];
  int n = blockIdx.x, tid = threadIdx.x;
  float bsum = bas[0] + bad[0];
  if (tid < 32) sdl[tid] = sdstv[n*32 + tid];
  for (int j = tid; j < 2048; j += 256) accs[j] = 0.f;
  __syncthreads();
  int beg = offsets[n], cnt = offsets[n+1] - beg;
  int t = tid >> 3, s8 = tid & 7;                 // 8 lanes per t, groups within a wave
  float base = sdl[t] + bsum;
  // pass 1: segment max of leaky logits
  float mx = -3.4e38f;
  for (int q = s8; q < cnt; q += 8){
    int e = elist[beg + q];
    float l = eattr[e]*ssrc[esrc[e]*32 + t] + base;
    l = l > 0.f ? l : SLOPE*l;
    mx = fmaxf(mx, l);
  }
  mx = fmaxf(mx, __shfl_xor(mx, 1));
  mx = fmaxf(mx, __shfl_xor(mx, 2));
  mx = fmaxf(mx, __shfl_xor(mx, 4));
  if (s8 == 0) smax[t] = mx;
  __syncthreads();
  // pass 2: denom
  float M = smax[t];
  float sm = 0.f;
  for (int q = s8; q < cnt; q += 8){
    int e = elist[beg + q];
    float l = eattr[e]*ssrc[esrc[e]*32 + t] + base;
    l = l > 0.f ? l : SLOPE*l;
    sm += __expf(l - M);
  }
  sm += __shfl_xor(sm, 1);
  sm += __shfl_xor(sm, 2);
  sm += __shfl_xor(sm, 4);
  if (s8 == 0) sden[t] = sm + 1e-16f;
  __syncthreads();
  // pass 3: accumulate att*attr*z[src] in LDS, chunked over edges
  for (int c0 = 0; c0 < cnt; c0 += CHUNK){
    int nch = min(CHUNK, cnt - c0);
    if (tid < nch){
      int e = elist[beg + c0 + tid];
      int s = esrc[e];
      sb[tid] = s;
      ab[tid] = eattr[e];
      __builtin_prefetch(z + (size_t)s*2048, 0, 2);  // pull z rows toward cache
    }
    __syncthreads();
    for (int q = tid; q < nch*32; q += 256){
      int ee = q >> 5, tt = q & 31;
      float l = ab[ee]*ssrc[sb[ee]*32 + tt] + sdl[tt] + bsum;
      l = l > 0.f ? l : SLOPE*l;
      cb[q] = __expf(l - smax[tt]) / sden[tt] * ab[ee];
    }
    __syncthreads();
    int h = tid & 63, tg = tid >> 6;
    for (int ee = 0; ee < nch; ++ee){
      const float* zr = z + (size_t)sb[ee]*2048;
      for (int tt = tg; tt < 32; tt += 4)
        accs[tt*64 + h] += cb[ee*32 + tt] * zr[tt*64 + h];
    }
    __syncthreads();
  }
  for (int j = tid; j < 2048; j += 256) out[(size_t)n*2048 + j] += accs[j];
}

// ---------------------------------------------------------------------- host
template <typename Tt>
static Tt* carve(char*& p, size_t n){
  Tt* r = (Tt*)p;
  size_t b = (n*sizeof(Tt) + 255) & ~(size_t)255;
  p += b;
  return r;
}

extern "C" void kernel_launch(void* const* d_in, const int* in_sizes, int n_in,
                              void* d_out, int out_size, void* d_ws, size_t ws_size,
                              hipStream_t stream) {
  (void)n_in; (void)out_size; (void)ws_size;
  const float* x     = (const float*)d_in[0];
  const int*   fei   = (const int*)  d_in[1];
  const float* fea   = (const float*)d_in[2];
  const float* Wproj = (const float*)d_in[3];
  const float* mix   = (const float*)d_in[4];
  const float* cfw   = (const float*)d_in[5];
  const float* cfb   = (const float*)d_in[6];
  const float* csw   = (const float*)d_in[7];
  const float* csb   = (const float*)d_in[8];
  const float* sgate = (const float*)d_in[9];
  const float* Wmsg  = (const float*)d_in[10];
  const float* bmsg  = (const float*)d_in[11];
  const float* Wself = (const float*)d_in[12];
  const float* bself = (const float*)d_in[13];
  const float* was   = (const float*)d_in[14];
  const float* bas   = (const float*)d_in[15];
  const float* wad   = (const float*)d_in[16];
  const float* bad   = (const float*)d_in[17];

  const int N    = in_sizes[0] / (TT*CC);   // 4096
  const int E    = in_sizes[2];             // 65536
  const int Etot = E + N*KTOP;
  const int rows = N*TT;

  char* p = (char*)d_ws;
  unsigned short* xbp    = carve<unsigned short>(p, (size_t)N*PADR*CC);
  unsigned short* wconv  = carve<unsigned short>(p, 2*3*64*64);
  unsigned short* wmsgT  = carve<unsigned short>(p, 64*64);
  unsigned short* wselfT = carve<unsigned short>(p, 64*64);
  float*          bcomb  = carve<float>(p, 64);
  float*          ev     = carve<float>(p, (size_t)N*32);
  int*            esrc   = carve<int>(p, Etot);
  int*            edst   = carve<int>(p, Etot);
  float*          eattr  = carve<float>(p, Etot);
  float*          xtf    = carve<float>(p, (size_t)rows*HH);
  unsigned short* xth    = carve<unsigned short>(p, (size_t)rows*HH);
  float*          zf     = carve<float>(p, (size_t)rows*HH);
  float*          ssrc   = carve<float>(p, rows);
  float*          sdstv  = carve<float>(p, rows);
  int*            cnts   = carve<int>(p, 2*N);      // counts then cursor, zeroed together
  int*            offs   = carve<int>(p, N+1);
  int*            elist  = carve<int>(p, Etot);
  int* counts = cnts;
  int* cursor = cnts + N;

  k_prep<<<1, 256, 0, stream>>>(cfw, cfb, csw, csb, sgate, Wmsg, Wself,
                                wconv, wmsgT, wselfT, bcomb);
  int totp = N*PADR*CC;
  k_castpad<<<(totp + 255)/256, 256, 0, stream>>>(x, xbp, totp);
  k_meanproj<<<N, 64, 0, stream>>>(x, Wproj, ev, N);
  k_topk<<<N, 256, 0, stream>>>(ev, mix, esrc, edst, eattr, N, E);
  k_fixed<<<(E + 255)/256, 256, 0, stream>>>(fei, fea, mix, esrc, edst, eattr, E);
  k_conv<<<(N*64)/256, 256, 0, stream>>>(xbp, wconv, bcomb, xtf, xth, N);
  int gemmBlocks = ((rows/16)*32)/256;
  k_gemm<<<gemmBlocks, 256, 0, stream>>>(xth, 32, 0, wmsgT, bmsg, zf, rows);
  k_gemm<<<gemmBlocks, 256, 0, stream>>>(xbp, PADR, PADO, wselfT, bself, (float*)d_out, rows);
  k_scores<<<(rows + 255)/256, 256, 0, stream>>>(zf, xtf, was, wad, ssrc, sdstv, rows);
  k_zero<<<(2*N + 255)/256, 256, 0, stream>>>(cnts, 2*N);
  k_count<<<(Etot + 255)/256, 256, 0, stream>>>(edst, counts, Etot);
  k_scan<<<1, 1, 0, stream>>>(counts, offs, N);
  k_fill<<<(Etot + 255)/256, 256, 0, stream>>>(edst, offs, cursor, elist, Etot);
  k_agg<<<N, 256, 0, stream>>>(esrc, eattr, offs, elist, ssrc, sdstv, zf, bas, bad, (float*)d_out);
}
